// MiniMaxM2Attention_66571993088732
// MI455X (gfx1250) — matrix operand order repaired and checked by reading, not compiled, not run on hardware
//
#include <hip/hip_runtime.h>

typedef __attribute__((ext_vector_type(16))) __bf16 bf16x16;
typedef __attribute__((ext_vector_type(8)))  float  f32x8;

constexpr int S   = 2048;
constexpr int HID = 4096;
constexpr int H   = 32;
constexpr int KV  = 8;
constexpr int D   = 128;
constexpr int NQ  = H * D;          // 4096
constexpr int NK  = KV * D;         // 1024
constexpr int NQKV = NQ + 2 * NK;   // 6144
constexpr float EPS = 1e-6f;

// round-to-nearest-even f32 -> bf16 (integer path; only used OUTSIDE matmul loops)
__device__ __forceinline__ __bf16 f2bf(float f) {
  union { float f; unsigned int u; } in; in.f = f;
  unsigned int r = in.u + 0x7FFFu + ((in.u >> 16) & 1u);
  union { unsigned short s; __bf16 b; } out; out.s = (unsigned short)(r >> 16);
  return out.b;
}

union frag16 { uint4 u[2]; bf16x16 v; };   // 32B fragment bit-cast

__device__ __forceinline__ unsigned lds_addr32(const void* p) {
  // generic pointer to LDS: low 32 bits are the LDS byte address (ISA 10.2)
  return (unsigned)(size_t)p;
}

__device__ __forceinline__ void async_ld_b128(unsigned lds_off, const __bf16* g) {
  asm volatile("global_load_async_to_lds_b128 %0, %1, off"
               :: "v"(lds_off), "v"(g) : "memory");
}

__device__ __forceinline__ void wait_async0() {
  asm volatile("s_wait_asynccnt 0x0" ::: "memory");
}

// ---------------------------------------------------------------------------
// One-pass f32 -> bf16 conversion (vectorized), n divisible by 4
// ---------------------------------------------------------------------------
__global__ __launch_bounds__(256)
void f32_to_bf16(const float* __restrict__ src, __bf16* __restrict__ dst, int n)
{
  const int stride = gridDim.x * blockDim.x * 4;
  for (int i = (blockIdx.x * blockDim.x + threadIdx.x) * 4; i < n; i += stride) {
    float4 v = *(const float4*)(src + i);
    union { __bf16 b[4]; uint2 u; } o;
    o.b[0] = f2bf(v.x); o.b[1] = f2bf(v.y);
    o.b[2] = f2bf(v.z); o.b[3] = f2bf(v.w);
    *(uint2*)(dst + i) = o.u;
  }
}

// ---------------------------------------------------------------------------
// bf16 GEMM: C[m, coff+n] = sum_k A[m,k] * B[n,k]   (A,B bf16 row-major)
// Workgroup (2 waves) -> 32(M) x 128(N). A tile (32x32) staged in LDS via
// double-buffered global_load_async_to_lds_b128; B streamed direct to VGPRs.
// grid = (N/128, M/32), block = 64
// ---------------------------------------------------------------------------
__global__ __launch_bounds__(64)
void gemm_bf16(const __bf16* __restrict__ A, const __bf16* __restrict__ B,
               float* __restrict__ C, int K, int ldc, int coff)
{
  __shared__ __bf16 atile[2][32 * 32];     // 2 x 2 KB, row stride 32 elems

  const int tid  = threadIdx.x;
  const int lane = tid & 31;
  const int wave = tid >> 5;               // 0..1
  const int l15  = lane & 15;
  const int lhi  = lane >> 4;              // 0/1
  const int rowbase = blockIdx.y * 32;
  const int colbase = blockIdx.x * 128 + wave * 64;

  f32x8 acc[8] = {};

  const __bf16* Brow = B + (size_t)(colbase + l15) * K;

  // async-copy assignment: thread tid copies 16B chunks (m = tid/4, c = tid%4)
  // and (m+16, c) of the 32-row x 32-col (64B/row) A tile.
  const int cm = tid >> 2;                 // 0..15
  const int cc = tid & 3;                  // 0..3
  const __bf16* Ag0 = A + (size_t)(rowbase + cm) * K + cc * 8;
  const __bf16* Ag1 = A + (size_t)(rowbase + 16 + cm) * K + cc * 8;
  const unsigned loff0 = (unsigned)(cm * 64 + cc * 16);
  const unsigned loff1 = (unsigned)((16 + cm) * 64 + cc * 16);

  const unsigned lbase0 = lds_addr32(&atile[0][0]);
  const unsigned lbase1 = lds_addr32(&atile[1][0]);

  // prologue: stage tile 0
  async_ld_b128(lbase0 + loff0, Ag0);
  async_ld_b128(lbase0 + loff1, Ag1);
  wait_async0();
  __syncthreads();

  const int niter = K / 32;
  for (int i = 0; i < niter; ++i) {
    const int k0 = i * 32;
    // prefetch next A tile into the other buffer (DMA overlaps WMMAs below)
    if (i + 1 < niter) {
      const unsigned nb = ((i + 1) & 1) ? lbase1 : lbase0;
      async_ld_b128(nb + loff0, Ag0 + k0 + 32);
      async_ld_b128(nb + loff1, Ag1 + k0 + 32);
    }

    const __bf16* abuf = &atile[i & 1][0];
    frag16 a0, a1;
    a0.u[0] = *(const uint4*)(abuf + l15 * 32 + 16 * lhi);
    a0.u[1] = *(const uint4*)(abuf + l15 * 32 + 16 * lhi + 8);
    a1.u[0] = *(const uint4*)(abuf + (16 + l15) * 32 + 16 * lhi);
    a1.u[1] = *(const uint4*)(abuf + (16 + l15) * 32 + 16 * lhi + 8);

    #pragma unroll
    for (int t = 0; t < 4; ++t) {
      const __bf16* Bp = Brow + (size_t)t * 16 * K + k0 + 16 * lhi;
      frag16 b;
      b.u[0] = *(const uint4*)(Bp);
      b.u[1] = *(const uint4*)(Bp + 8);
      acc[t] = __builtin_amdgcn_wmma_f32_16x16x32_bf16(
          false, a0.v, false, b.v, (short)0, acc[t], false, false);
      acc[4 + t] = __builtin_amdgcn_wmma_f32_16x16x32_bf16(
          false, a1.v, false, b.v, (short)0, acc[4 + t], false, false);
    }

    // this wave's DMA done + all waves' DMA visible before next tile is read
    wait_async0();
    __syncthreads();
  }

  // store C/D layout: lane holds col l15, rows r + 8*lhi (two 16-row halves)
  #pragma unroll
  for (int t = 0; t < 4; ++t) {
    const int n = colbase + t * 16 + l15;
    #pragma unroll
    for (int r = 0; r < 8; ++r) {
      C[(size_t)(rowbase + r + 8 * lhi) * ldc + coff + n]      = acc[t][r];
      C[(size_t)(rowbase + 16 + r + 8 * lhi) * ldc + coff + n] = acc[4 + t][r];
    }
  }
}

// ---------------------------------------------------------------------------
// RMSNorm (q over 4096, k over 1024) + partial RoPE (R=64, pairs (d, d+32)),
// emit bf16 Q [S][H*D], bf16 K [S][KV*D], bf16 V^T [KV][D][S].
// grid = S blocks, 256 threads
// ---------------------------------------------------------------------------
__global__ __launch_bounds__(256)
void norm_rope(const float* __restrict__ qkv,
               const float* __restrict__ qnw, const float* __restrict__ knw,
               __bf16* __restrict__ qb, __bf16* __restrict__ kb,
               __bf16* __restrict__ vbT)
{
  __shared__ float red[256];
  __shared__ float srq, srk;
  const int s   = blockIdx.x;
  const int tid = threadIdx.x;
  const float* row = qkv + (size_t)s * NQKV;

  float acc = 0.f;
  for (int i = tid; i < NQ; i += 256) { float v = row[i]; acc += v * v; }
  red[tid] = acc; __syncthreads();
  for (int off = 128; off > 0; off >>= 1) {
    if (tid < off) red[tid] += red[tid + off];
    __syncthreads();
  }
  if (tid == 0) srq = rsqrtf(red[0] * (1.0f / NQ) + EPS);
  __syncthreads();

  float acck = 0.f;
  for (int i = tid; i < NK; i += 256) { float v = row[NQ + i]; acck += v * v; }
  red[tid] = acck; __syncthreads();
  for (int off = 128; off > 0; off >>= 1) {
    if (tid < off) red[tid] += red[tid + off];
    __syncthreads();
  }
  if (tid == 0) srk = rsqrtf(red[0] * (1.0f / NK) + EPS);
  __syncthreads();

  const float rq = srq, rk = srk;
  const float pos = (float)s;   // position_ids == arange(S)

  for (int i = tid; i < NQ; i += 256) {
    const int d = i & (D - 1);
    float v = row[i] * rq * qnw[i];
    float outv;
    if (d < 32) {
      const float fr = pos * __powf(10000.f, -(float)d * (1.f / 32.f));
      const float v2 = row[i + 32] * rq * qnw[i + 32];
      outv = v * __cosf(fr) - v2 * __sinf(fr);
    } else if (d < 64) {
      const float fr = pos * __powf(10000.f, -(float)(d - 32) * (1.f / 32.f));
      const float v1 = row[i - 32] * rq * qnw[i - 32];
      outv = v * __cosf(fr) + v1 * __sinf(fr);
    } else {
      outv = v;
    }
    qb[(size_t)s * NQ + i] = f2bf(outv);
  }
  for (int i = tid; i < NK; i += 256) {
    const int d = i & (D - 1);
    float v = row[NQ + i] * rk * knw[i];
    float outv;
    if (d < 32) {
      const float fr = pos * __powf(10000.f, -(float)d * (1.f / 32.f));
      const float v2 = row[NQ + i + 32] * rk * knw[i + 32];
      outv = v * __cosf(fr) - v2 * __sinf(fr);
    } else if (d < 64) {
      const float fr = pos * __powf(10000.f, -(float)(d - 32) * (1.f / 32.f));
      const float v1 = row[NQ + i - 32] * rk * knw[i - 32];
      outv = v * __cosf(fr) + v1 * __sinf(fr);
    } else {
      outv = v;
    }
    kb[(size_t)s * NK + i] = f2bf(outv);
  }
  for (int i = tid; i < NK; i += 256) {
    const int kv = i >> 7, d = i & (D - 1);
    vbT[((size_t)kv * D + d) * S + s] = f2bf(row[NQ + NK + i]);
  }
}

// ---------------------------------------------------------------------------
// Flash attention, one wave per (16-query tile, head). 32-key blocks.
// Writes bf16 output so the out-proj GEMM consumes it directly.
// grid = (S/16, H), block = 32
// ---------------------------------------------------------------------------
__global__ __launch_bounds__(32)
void attention(const __bf16* __restrict__ qb, const __bf16* __restrict__ kb,
               const __bf16* __restrict__ vbT, __bf16* __restrict__ attnb)
{
  __shared__ float  sc[16][32];
  __shared__ __bf16 pbuf[16][32];
  __shared__ float  salpha[16];
  __shared__ float  slinv[16];

  const int lane = threadIdx.x & 31;
  const int l15  = lane & 15;
  const int lhi  = lane >> 4;
  const int qtile = blockIdx.x;
  const int h     = blockIdx.y;
  const int kvh   = h >> 2;               // H/KV = 4
  const int qbase = qtile * 16;
  const float scale = 0.08838834764831845f;  // 128^-0.5

  // Q fragments: 4 x (16x32 bf16) A-layout
  bf16x16 qf[4];
  {
    const __bf16* qrow = qb + (size_t)(qbase + l15) * NQ + h * D;
    #pragma unroll
    for (int f = 0; f < 4; ++f) {
      frag16 q;
      q.u[0] = *(const uint4*)(qrow + 32 * f + 16 * lhi);
      q.u[1] = *(const uint4*)(qrow + 32 * f + 16 * lhi + 8);
      qf[f] = q.v;
    }
  }

  f32x8 o[8] = {};
  float rowm = -3.0e38f, rowl = 0.f;   // live in lanes 0..15

  const int nblocks = (qbase + 16 + 31) / 32;
  for (int blk = 0; blk < nblocks; ++blk) {
    const int kbase = blk * 32;

    // ---- scores: two 16-key subtiles, K=128 via 4 WMMAs each
    #pragma unroll
    for (int t = 0; t < 2; ++t) {
      f32x8 sacc = {};
      const int key = kbase + 16 * t + l15;
      const __bf16* krow = kb + (size_t)key * NK + kvh * D + 16 * lhi;
      #pragma unroll
      for (int f = 0; f < 4; ++f) {
        frag16 b;
        b.u[0] = *(const uint4*)(krow + 32 * f);
        b.u[1] = *(const uint4*)(krow + 32 * f + 8);
        sacc = __builtin_amdgcn_wmma_f32_16x16x32_bf16(
            false, qf[f], false, b.v, (short)0, sacc, false, false);
      }
      #pragma unroll
      for (int r = 0; r < 8; ++r)
        sc[r + 8 * lhi][16 * t + l15] = sacc[r] * scale;
    }
    __syncthreads();

    // ---- online softmax: lanes 0..15 each own a query row
    if (lane < 16) {
      const int qg = qbase + lane;
      float m = rowm;
      float v[32];
      #pragma unroll
      for (int c = 0; c < 32; ++c) {
        float x = sc[lane][c];
        if (kbase + c > qg) x = -3.0e38f;   // causal mask
        v[c] = x;
        m = fmaxf(m, x);
      }
      const float alpha = __expf(rowm - m);
      float sum = 0.f;
      #pragma unroll
      for (int c = 0; c < 32; ++c) {
        const float p = __expf(v[c] - m);
        sum += p;
        pbuf[lane][c] = f2bf(p);
      }
      rowl = rowl * alpha + sum;
      rowm = m;
      salpha[lane] = alpha;
    }
    __syncthreads();

    // ---- rescale O accumulators
    float al[8];
    #pragma unroll
    for (int r = 0; r < 8; ++r) al[r] = salpha[r + 8 * lhi];
    #pragma unroll
    for (int ch = 0; ch < 8; ++ch)
      #pragma unroll
      for (int r = 0; r < 8; ++r)
        o[ch][r] *= al[r];

    // ---- P fragment (A-layout 16x32) from LDS
    bf16x16 pf;
    {
      frag16 p;
      p.u[0] = *(const uint4*)(&pbuf[l15][16 * lhi]);
      p.u[1] = *(const uint4*)(&pbuf[l15][16 * lhi + 8]);
      pf = p.v;
    }

    // ---- O += P * V  (8 WMMAs over D=128)
    #pragma unroll
    for (int ch = 0; ch < 8; ++ch) {
      const int d = 16 * ch + l15;
      const __bf16* vrow = vbT + ((size_t)kvh * D + d) * S + kbase + 16 * lhi;
      frag16 b;
      b.u[0] = *(const uint4*)(vrow);
      b.u[1] = *(const uint4*)(vrow + 8);
      o[ch] = __builtin_amdgcn_wmma_f32_16x16x32_bf16(
          false, pf, false, b.v, (short)0, o[ch], false, false);
    }
    __syncthreads();
  }

  // ---- finalize: divide by row sums, write bf16 [S][H*D]
  if (lane < 16) slinv[lane] = 1.0f / rowl;
  __syncthreads();
  float li[8];
  #pragma unroll
  for (int r = 0; r < 8; ++r) li[r] = slinv[r + 8 * lhi];
  #pragma unroll
  for (int ch = 0; ch < 8; ++ch)
    #pragma unroll
    for (int r = 0; r < 8; ++r)
      attnb[(size_t)(qbase + r + 8 * lhi) * NQ + h * D + 16 * ch + l15] =
          f2bf(o[ch][r] * li[r]);
}

// ---------------------------------------------------------------------------
extern "C" void kernel_launch(void* const* d_in, const int* in_sizes, int n_in,
                              void* d_out, int out_size, void* d_ws, size_t ws_size,
                              hipStream_t stream) {
  (void)in_sizes; (void)n_in; (void)out_size; (void)ws_size;
  const float* x   = (const float*)d_in[0];
  const float* Wq  = (const float*)d_in[1];
  const float* Wk  = (const float*)d_in[2];
  const float* Wv  = (const float*)d_in[3];
  const float* Wo  = (const float*)d_in[4];
  const float* qnw = (const float*)d_in[5];
  const float* knw = (const float*)d_in[6];
  // d_in[7] (position_ids) == arange(S); positions derived analytically

  char* ws = (char*)d_ws;
  size_t off = 0;
  __bf16* xb   = (__bf16*)(ws + off); off += (size_t)S * HID * 2;     // 16.8 MB
  __bf16* Wqb  = (__bf16*)(ws + off); off += (size_t)NQ * HID * 2;    // 33.6 MB
  __bf16* Wkb  = (__bf16*)(ws + off); off += (size_t)NK * HID * 2;    //  8.4 MB
  __bf16* Wvb  = (__bf16*)(ws + off); off += (size_t)NK * HID * 2;    //  8.4 MB
  __bf16* Wob  = (__bf16*)(ws + off); off += (size_t)HID * NQ * 2;    // 33.6 MB
  float*  qkv  = (float*)(ws + off);  off += (size_t)S * NQKV * 4;    // 50.3 MB
  __bf16* qb   = (__bf16*)(ws + off); off += (size_t)S * NQ * 2;      // 16.8 MB
  __bf16* kbp  = (__bf16*)(ws + off); off += (size_t)S * NK * 2;      //  4.2 MB
  __bf16* vbT  = (__bf16*)(ws + off); off += (size_t)KV * D * S * 2;  //  4.2 MB
  __bf16* attn = (__bf16*)(ws + off); off += (size_t)S * NQ * 2;      // 16.8 MB
  float*  out  = (float*)d_out;

  // One-pass fp32 -> bf16 conversions (hoisted out of all matmul loops)
  f32_to_bf16<<<1024, 256, 0, stream>>>(x,  xb,  S * HID);
  f32_to_bf16<<<1024, 256, 0, stream>>>(Wq, Wqb, NQ * HID);
  f32_to_bf16<<<1024, 256, 0, stream>>>(Wk, Wkb, NK * HID);
  f32_to_bf16<<<1024, 256, 0, stream>>>(Wv, Wvb, NK * HID);
  f32_to_bf16<<<1024, 256, 0, stream>>>(Wo, Wob, HID * NQ);

  // QKV projections (fused f32 output buffer, ldc = 6144)
  gemm_bf16<<<dim3(NQ / 128, S / 32), 64, 0, stream>>>(xb, Wqb, qkv, HID, NQKV, 0);
  gemm_bf16<<<dim3(NK / 128, S / 32), 64, 0, stream>>>(xb, Wkb, qkv, HID, NQKV, NQ);
  gemm_bf16<<<dim3(NK / 128, S / 32), 64, 0, stream>>>(xb, Wvb, qkv, HID, NQKV, NQ + NK);

  // RMSNorm + RoPE + bf16 pack (V transposed for contiguous PV B-operand)
  norm_rope<<<S, 256, 0, stream>>>(qkv, qnw, knw, qb, kbp, vbT);

  // Flash attention (bf16 output)
  attention<<<dim3(S / 16, H), 32, 0, stream>>>(qb, kbp, vbT, attn);

  // Output projection -> fp32 d_out
  gemm_bf16<<<dim3(HID / 128, S / 32), 64, 0, stream>>>(attn, Wob, out, NQ, HID, 0);
}
